// catcher_47201690583536
// MI455X (gfx1250) — matrix operand order repaired and checked by reading, not compiled, not run on hardware
//
#include <hip/hip_runtime.h>
#include <math.h>

typedef __attribute__((ext_vector_type(16))) __bf16 v16bf;
typedef __attribute__((ext_vector_type(8)))  float  v8f;

#define BDIM 8
#define NDIM 1920
#define CDIM 512
#define BLK_TOK 192
#define NROWS (BDIM*NDIM)        // 15360
#define NBLK  (NROWS/BLK_TOK)    // 80
#define SCALE_F 0.022097086912079612f  // 2048^-0.5

#define LDS_PITCH 72             // halfs per row: 64 + 8 pad -> 144B rows, 16B aligned

// ---- gfx1250 async load-to-LDS (ASYNCcnt path), with sync fallback ----
#if defined(__has_builtin)
#  if __has_builtin(__builtin_amdgcn_global_load_async_to_lds_b128)
#    define USE_ASYNC_LDS 1
#  endif
#endif

// builtin parameter types: pointers to 4xi32 GCC-vector, AS(1) src / AS(3) dst
typedef int v4i_g __attribute__((vector_size(16)));
typedef __attribute__((address_space(1))) v4i_g* gv4p;
typedef __attribute__((address_space(3))) v4i_g* lv4p;

// as(1) pointer value == generic global address (integer round-trip)
__device__ __forceinline__ gv4p as_glb(const void* p) {
    return (gv4p)(unsigned long long)p;
}
// as(3) pointer value == low 32 bits of the generic LDS address
__device__ __forceinline__ lv4p as_lds(const void* p) {
    return (lv4p)(unsigned)(unsigned long long)p;
}

__device__ __forceinline__ void wait_async0() {
#if defined(USE_ASYNC_LDS)
#  if __has_builtin(__builtin_amdgcn_s_wait_asynccnt)
    __builtin_amdgcn_s_wait_asynccnt(0);
#  else
    asm volatile("s_wait_asynccnt 0x0" ::: "memory");
#  endif
#endif
}

__device__ __forceinline__ unsigned short f2bf(float f) {
    unsigned int u = __float_as_uint(f);
    unsigned int r = u + 0x7FFFu + ((u >> 16) & 1u);   // round-to-nearest-even
    return (unsigned short)(r >> 16);
}

// ---------------- mean over tokens: gmean[b,c] ----------------
__global__ __launch_bounds__(256) void mean_kernel(const float* __restrict__ g,
                                                   float* __restrict__ gmean) {
    int t = blockIdx.x * 256 + threadIdx.x;        // 0 .. B*C-1
    int b = t / CDIM, c = t % CDIM;
    const float* p = g + (size_t)b * NDIM * CDIM + c;
    float s = 0.f;
    for (int n = 0; n < NDIM; ++n) s += p[(size_t)n * CDIM];
    gmean[t] = s * (1.0f / NDIM);
}

// ---------------- f32 -> bf16 convert ----------------
__global__ __launch_bounds__(256) void cvt_kernel(const float* __restrict__ src,
                                                  unsigned short* __restrict__ dst, int n) {
    int i = blockIdx.x * 256 + threadIdx.x;
    if (i < n) dst[i] = f2bf(src[i]);
}

// ---------------- per-row prep: relbf (normalized g), qbf (x+gmean), gT ----------------
__global__ __launch_bounds__(256) void prep_rows(const float* __restrict__ g,
                                                 const float* __restrict__ x,
                                                 const float* __restrict__ gmean,
                                                 unsigned short* __restrict__ relbf,
                                                 unsigned short* __restrict__ qbf,
                                                 unsigned short* __restrict__ gT) {
    int lane = threadIdx.x & 31;
    int wave = threadIdx.x >> 5;
    int row  = blockIdx.x * 8 + wave;              // 0..15359
    int b    = row / NDIM;
    int blk  = row / BLK_TOK;
    int j    = row % BLK_TOK;
    const float* gr = g + (size_t)row * CDIM;
    const float* xr = x + (size_t)row * CDIM;
    const float* gm = gmean + (size_t)b * CDIM;

    float v[16];
    float ss = 0.f;
    #pragma unroll
    for (int i = 0; i < 16; ++i) { int c = i * 32 + lane; v[i] = gr[c]; ss += v[i] * v[i]; }
    #pragma unroll
    for (int off = 16; off >= 1; off >>= 1) ss += __shfl_xor(ss, off, 32);
    float inv = 1.0f / fmaxf(sqrtf(ss), 1e-12f);

    unsigned short* gTb = gT + (size_t)blk * CDIM * BLK_TOK + j;
    #pragma unroll
    for (int i = 0; i < 16; ++i) {
        int c = i * 32 + lane;
        relbf[(size_t)row * CDIM + c] = f2bf(v[i] * inv);
        qbf  [(size_t)row * CDIM + c] = f2bf(xr[c] + gm[c]);
        gTb  [(size_t)c * BLK_TOK]    = f2bf(v[i]);
    }
}

// ---------------- row softmax: f32 scores -> bf16 probs ----------------
__global__ __launch_bounds__(256) void softmax_rows(const float* __restrict__ S,
                                                    unsigned short* __restrict__ P, int L) {
    __shared__ float red[256];
    int row = blockIdx.x;
    const float* s = S + (size_t)row * L;
    unsigned short* p = P + (size_t)row * L;
    int t = threadIdx.x;
    float m = -1e30f;
    for (int i = t; i < L; i += 256) m = fmaxf(m, s[i]);
    red[t] = m; __syncthreads();
    for (int o = 128; o > 0; o >>= 1) { if (t < o) red[t] = fmaxf(red[t], red[t + o]); __syncthreads(); }
    m = red[0]; __syncthreads();
    float sum = 0.f;
    for (int i = t; i < L; i += 256) sum += __expf(s[i] - m);
    red[t] = sum; __syncthreads();
    for (int o = 128; o > 0; o >>= 1) { if (t < o) red[t] += red[t + o]; __syncthreads(); }
    float inv = 1.0f / red[0];
    for (int i = t; i < L; i += 256) p[i] = f2bf(__expf(s[i] - m) * inv);
}

// ---------------- generic C = A * Bm^T WMMA GEMM ----------------
// C[m,n] = sum_k A[m,k]*Bm[n,k]. Per-z offsets: (z/div)*outer + (z%div)*inner.
// 128 threads = 4 waves (2x2), wave tile 32x32 (2x2 of 16x16), block tile 64x64,
// K-step 64 -> 8 v_wmma per wave per barrier pair; staging via async load-to-LDS.
struct GemmP {
    const unsigned short* A; const unsigned short* B;
    float* Cf; unsigned short* Cb; unsigned short* CT;
    int lda, ldb, ldc, ldt;
    int K;
    float alpha;
    long long aOut, aIn; int aDiv;
    long long bOut, bIn; int bDiv;
    long long cOut, cIn; int cDiv;
    long long tOut, tIn; int tDiv;
};

__device__ __forceinline__ long long zoff(int z, int dv, long long outer, long long inner) {
    return (long long)(z / dv) * outer + (long long)(z % dv) * inner;
}

union FragU { uint4 q[2]; v16bf v; };

__device__ __forceinline__ void loadA(FragU& f, const unsigned short* As,
                                      int row, int kk, int g16) {
    const unsigned short* p = &As[row * LDS_PITCH + kk * 32 + g16 * 8];
    f.q[0] = *reinterpret_cast<const uint4*>(p);
    f.q[1] = *reinterpret_cast<const uint4*>(p + 16);
}
__device__ __forceinline__ void loadB(FragU& f, const unsigned short* Bs,
                                      int row, int kk, int g16) {
    const unsigned short* p = &Bs[row * LDS_PITCH + kk * 32 + g16 * 8];
    f.q[0] = *reinterpret_cast<const uint4*>(p);
    f.q[1] = *reinterpret_cast<const uint4*>(p + 16);
}

__global__ __launch_bounds__(128) void gemm_abT(GemmP P) {
    __shared__ unsigned short As[64 * LDS_PITCH];   // 64 x 64 tile
    __shared__ unsigned short Bs[64 * LDS_PITCH];   // 64 x 64 tile
    const int t    = threadIdx.x;
    const int lane = t & 31, wave = t >> 5;
    const int mw = wave & 1, nw = wave >> 1;        // 2x2 wave grid
    const int z = blockIdx.z;
    const int mBase = blockIdx.y * 64;
    const int nBase = blockIdx.x * 64;

    const unsigned short* Ag = P.A + zoff(z, P.aDiv, P.aOut, P.aIn) + (size_t)mBase * P.lda;
    const unsigned short* Bg = P.B + zoff(z, P.bDiv, P.bOut, P.bIn) + (size_t)nBase * P.ldb;

    const int g16 = lane >> 4;               // k-group select
    const int l16 = lane & 15;
    const int srow = t >> 1, scol = (t & 1) * 4;  // 128 threads: 64 rows x 2 halves (4x16B each)

    v8f acc00 = {}, acc01 = {}, acc10 = {}, acc11 = {};

    for (int k = 0; k < P.K; k += 64) {
        __syncthreads();
        // stage 64x64 A and B panels into LDS: 8x 16B chunks per thread
#if defined(USE_ASYNC_LDS)
        #pragma unroll
        for (int i = 0; i < 4; ++i) {
            int c = scol + i;                // chunk column 0..7
            __builtin_amdgcn_global_load_async_to_lds_b128(
                as_glb(Ag + (size_t)srow * P.lda + k + c * 8),
                as_lds(&As[srow * LDS_PITCH + c * 8]), 0, 0);
            __builtin_amdgcn_global_load_async_to_lds_b128(
                as_glb(Bg + (size_t)srow * P.ldb + k + c * 8),
                as_lds(&Bs[srow * LDS_PITCH + c * 8]), 0, 0);
        }
        wait_async0();
#else
        {   // batched sync fallback: all loads first, then all LDS stores
            uint4 va[4], vb[4];
            #pragma unroll
            for (int i = 0; i < 4; ++i) {
                int c = scol + i;
                va[i] = *reinterpret_cast<const uint4*>(Ag + (size_t)srow * P.lda + k + c * 8);
                vb[i] = *reinterpret_cast<const uint4*>(Bg + (size_t)srow * P.ldb + k + c * 8);
            }
            #pragma unroll
            for (int i = 0; i < 4; ++i) {
                int c = scol + i;
                *reinterpret_cast<uint4*>(&As[srow * LDS_PITCH + c * 8]) = va[i];
                *reinterpret_cast<uint4*>(&Bs[srow * LDS_PITCH + c * 8]) = vb[i];
            }
        }
#endif
        __syncthreads();

        #pragma unroll
        for (int kk = 0; kk < 2; ++kk) {
            FragU a0, a1, b0, b1;
            loadA(a0, As, mw * 32 + l16,      kk, g16);
            loadA(a1, As, mw * 32 + 16 + l16, kk, g16);
            loadB(b0, Bs, nw * 32 + l16,      kk, g16);
            loadB(b1, Bs, nw * 32 + 16 + l16, kk, g16);
            acc00 = __builtin_amdgcn_wmma_f32_16x16x32_bf16(false, a0.v, false, b0.v, (short)0, acc00, false, false);
            acc01 = __builtin_amdgcn_wmma_f32_16x16x32_bf16(false, a0.v, false, b1.v, (short)0, acc01, false, false);
            acc10 = __builtin_amdgcn_wmma_f32_16x16x32_bf16(false, a1.v, false, b0.v, (short)0, acc10, false, false);
            acc11 = __builtin_amdgcn_wmma_f32_16x16x32_bf16(false, a1.v, false, b1.v, (short)0, acc11, false, false);
        }
    }

    // epilogue: VGPR r -> row r + 8*g16 within 16x16 tile, col l16
    float*          cf = P.Cf ? P.Cf + zoff(z, P.cDiv, P.cOut, P.cIn) : nullptr;
    unsigned short* cb = P.Cb ? P.Cb + zoff(z, P.cDiv, P.cOut, P.cIn) : nullptr;
    unsigned short* ct = P.CT ? P.CT + zoff(z, P.tDiv, P.tOut, P.tIn) : nullptr;
    const v8f* accs[4] = { &acc00, &acc01, &acc10, &acc11 };
    #pragma unroll
    for (int mt = 0; mt < 2; ++mt) {
        #pragma unroll
        for (int nt = 0; nt < 2; ++nt) {
            const v8f& a = *accs[mt * 2 + nt];
            const int gm0 = mBase + mw * 32 + mt * 16 + (g16 << 3);
            const int gn  = nBase + nw * 32 + nt * 16 + l16;
            #pragma unroll
            for (int r = 0; r < 8; ++r) {
                float val = a[r] * P.alpha;
                int gm = gm0 + r;
                if (cf) cf[(size_t)gm * P.ldc + gn] = val;
                if (cb) cb[(size_t)gm * P.ldc + gn] = f2bf(val);
                if (ct) ct[(size_t)gn * P.ldt + gm] = f2bf(val);
            }
        }
    }
}

// ---------------- host orchestration ----------------
static inline size_t alignUp(size_t v, size_t a) { return (v + a - 1) / a * a; }

extern "C" void kernel_launch(void* const* d_in, const int* in_sizes, int n_in,
                              void* d_out, int out_size, void* d_ws, size_t ws_size,
                              hipStream_t stream) {
    const float* x  = (const float*)d_in[0];
    const float* g  = (const float*)d_in[1];
    const float* Wq = (const float*)d_in[2];
    const float* Wg = (const float*)d_in[3];
    float* out = (float*)d_out;

    // ---- carve workspace ----
    char* ws = (char*)d_ws;
    size_t off = 0;
    auto carve = [&](size_t bytes) { size_t o = alignUp(off, 256); off = o + bytes; return ws + o; };
    const size_t RC  = (size_t)NROWS * CDIM;          // 15360*512
    float*          gmean   = (float*)         carve(BDIM * CDIM * 4);
    unsigned short* Wqbf    = (unsigned short*)carve((size_t)CDIM * CDIM * 2);
    unsigned short* Wgbf    = (unsigned short*)carve((size_t)CDIM * CDIM * 2);
    unsigned short* qbf     = (unsigned short*)carve(RC * 2);
    unsigned short* relbf   = (unsigned short*)carve(RC * 2);
    unsigned short* gT      = (unsigned short*)carve(RC * 2);   // [80][512][192]
    unsigned short* relq    = (unsigned short*)carve(RC * 2);
    unsigned short* g2bf    = (unsigned short*)carve(RC * 2);
    unsigned short* g2T     = (unsigned short*)carve(RC * 2);   // [B][512][1920]
    unsigned short* relg    = (unsigned short*)carve(RC * 2);
    float*          sblk    = (float*)         carve((size_t)NBLK * BLK_TOK * BLK_TOK * 4);
    unsigned short* attnblk = (unsigned short*)carve((size_t)NBLK * BLK_TOK * BLK_TOK * 2);
    float*          s2      = (float*)         carve((size_t)NDIM * NDIM * 4);
    unsigned short* attn2   = (unsigned short*)carve((size_t)NDIM * NDIM * 2);

    // ---- prep ----
    mean_kernel<<<BDIM * CDIM / 256, 256, 0, stream>>>(g, gmean);
    cvt_kernel<<<CDIM * CDIM / 256, 256, 0, stream>>>(Wq, Wqbf, CDIM * CDIM);
    cvt_kernel<<<CDIM * CDIM / 256, 256, 0, stream>>>(Wg, Wgbf, CDIM * CDIM);
    prep_rows<<<NROWS / 8, 256, 0, stream>>>(g, x, gmean, relbf, qbf, gT);

    auto launch = [&](GemmP& p, int M, int N, int Z) {
        dim3 grid(N / 64, M / 64, Z);
        gemm_abT<<<grid, 128, 0, stream>>>(p);
    };

    // ---- rel_q = q @ Wq^T (bf16 out) ----
    {
        GemmP p{}; p.A = qbf; p.B = Wqbf; p.Cb = relq;
        p.lda = CDIM; p.ldb = CDIM; p.ldc = CDIM; p.K = CDIM; p.alpha = 1.f;
        p.aDiv = p.bDiv = p.cDiv = p.tDiv = 1;
        launch(p, NROWS, CDIM, 1);
    }
    // ---- block-local scores = relbf @ relbf^T * SCALE (f32 out), z = 80 blocks ----
    {
        GemmP p{}; p.A = relbf; p.B = relbf; p.Cf = sblk;
        p.lda = CDIM; p.ldb = CDIM; p.ldc = BLK_TOK; p.K = CDIM; p.alpha = SCALE_F;
        p.aDiv = p.bDiv = p.cDiv = p.tDiv = 1;
        p.aOut = (long long)BLK_TOK * CDIM; p.bOut = p.aOut;
        p.cOut = (long long)BLK_TOK * BLK_TOK;
        launch(p, BLK_TOK, BLK_TOK, NBLK);
    }
    softmax_rows<<<NBLK * BLK_TOK, 256, 0, stream>>>(sblk, attnblk, BLK_TOK);
    // ---- g2 = attn @ gb  via A=attn, Bm=gT; write g2 row-major + g2T per-batch ----
    {
        GemmP p{}; p.A = attnblk; p.B = gT; p.Cb = g2bf; p.CT = g2T;
        p.lda = BLK_TOK; p.ldb = BLK_TOK; p.ldc = CDIM; p.ldt = NDIM;
        p.K = BLK_TOK; p.alpha = 1.f;
        p.aDiv = p.bDiv = p.cDiv = 1; p.tDiv = NDIM / BLK_TOK;   // 10 blocks per batch
        p.aOut = (long long)BLK_TOK * BLK_TOK;
        p.bOut = (long long)CDIM * BLK_TOK;
        p.cOut = (long long)BLK_TOK * CDIM;
        p.tOut = (long long)CDIM * NDIM; p.tIn = BLK_TOK;
        launch(p, BLK_TOK, CDIM, NBLK);
    }
    // ---- rel_g = g2 @ Wg^T (bf16 out) ----
    {
        GemmP p{}; p.A = g2bf; p.B = Wgbf; p.Cb = relg;
        p.lda = CDIM; p.ldb = CDIM; p.ldc = CDIM; p.K = CDIM; p.alpha = 1.f;
        p.aDiv = p.bDiv = p.cDiv = p.tDiv = 1;
        launch(p, NROWS, CDIM, 1);
    }
    // ---- global attention per batch (reuse s2/attn2 buffers to keep ws small) ----
    const long long BATCH = (long long)NDIM * CDIM;   // 983040
    for (int b = 0; b < BDIM; ++b) {
        {   // scores2 = rel_q @ rel_g^T * SCALE (f32)
            GemmP p{}; p.A = relq + b * BATCH; p.B = relg + b * BATCH; p.Cf = s2;
            p.lda = CDIM; p.ldb = CDIM; p.ldc = NDIM; p.K = CDIM; p.alpha = SCALE_F;
            p.aDiv = p.bDiv = p.cDiv = p.tDiv = 1;
            launch(p, NDIM, NDIM, 1);
        }
        softmax_rows<<<NDIM, 256, 0, stream>>>(s2, attn2, NDIM);
        {   // out_b = attn2 @ g2  via Bm = g2T (f32 out, final)
            GemmP p{}; p.A = attn2; p.B = g2T + b * BATCH; p.Cf = out + b * BATCH;
            p.lda = NDIM; p.ldb = NDIM; p.ldc = CDIM; p.K = NDIM; p.alpha = 1.f;
            p.aDiv = p.bDiv = p.cDiv = p.tDiv = 1;
            launch(p, NDIM, CDIM, 1);
        }
    }
}
